// DescrptSeT_45062796870087
// MI455X (gfx1250) — matrix operand order repaired and checked by reading, not compiled, not run on hardware
//
#include <hip/hip_runtime.h>
#include <hip/hip_bf16.h>
#include <math.h>

typedef __attribute__((ext_vector_type(16))) _Float16 v16h;
typedef __attribute__((ext_vector_type(8)))  float    v8f;

static constexpr int kNLOC = 512;
static constexpr int kNNEI = 60;
static constexpr int kNG   = 96;

__device__ __forceinline__ float fast_tanh(float x) {
#if __has_builtin(__builtin_amdgcn_tanhf)
  return __builtin_amdgcn_tanhf(x);   // v_tanh_f32 on gfx1250
#else
  return tanhf(x);
#endif
}

// ---------------- Kernel 1: environment matrix + switch function ----------
__global__ void env_mat_kernel(const int* __restrict__ nlist,
                               const float* __restrict__ coord,
                               const int* __restrict__ atype,
                               const float* __restrict__ mean,
                               const float* __restrict__ stddev,
                               float* __restrict__ dm,
                               float* __restrict__ sw_out) {
  int t = blockIdx.x * blockDim.x + threadIdx.x;
  if (t >= kNLOC * kNNEI) return;
  int i = t / kNNEI;
  int j = t % kNNEI;
  int n = nlist[t];
  bool mask = n >= 0;
  int nl = mask ? n : 0;
  float dx = coord[nl * 3 + 0] - coord[i * 3 + 0];
  float dy = coord[nl * 3 + 1] - coord[i * 3 + 1];
  float dz = coord[nl * 3 + 2] - coord[i * 3 + 2];
  float len = sqrtf(dx * dx + dy * dy + dz * dz);
  len += mask ? 0.f : 1.f;
  // smooth weight, rmin=0.5, rmax=6.0
  float u = (len - 0.5f) * (1.0f / 5.5f);
  float vv = u * u * u * (-6.0f * u * u + 15.0f * u - 10.0f) + 1.0f;
  float sw = (len <= 0.5f) ? 1.0f : ((len >= 6.0f) ? 0.0f : vv);
  sw = mask ? sw : 0.0f;
  float inv = 1.0f / len;
  float inv2 = inv * inv;
  float e0 = inv * sw;
  float e1 = dx * inv2 * sw;
  float e2 = dy * inv2 * sw;
  float e3 = dz * inv2 * sw;
  int at = atype[i];
  const float* mu = mean   + (at * kNNEI + j) * 4;
  const float* sd = stddev + (at * kNNEI + j) * 4;
  dm[t * 4 + 0] = (e0 - mu[0]) / sd[0];
  dm[t * 4 + 1] = (e1 - mu[1]) / sd[1];
  dm[t * 4 + 2] = (e2 - mu[2]) / sd[2];
  dm[t * 4 + 3] = (e3 - mu[3]) / sd[3];
  sw_out[t] = sw;
}

// ---------------- Kernel 2: three-body embedding via WMMA -----------------
// Block = one local atom, 256 threads = 8 waves; each wave owns 16-row tiles.
__launch_bounds__(256)
__global__ void three_body_kernel(const float* __restrict__ dm,
                                  const float* __restrict__ w0g,
                                  const float* __restrict__ b0g,
                                  const float* __restrict__ w1g,
                                  const float* __restrict__ b1g,
                                  const float* __restrict__ w2g,
                                  const float* __restrict__ b2g,
                                  float* __restrict__ out) {
  constexpr int H1S = 50;                    // padded LDS stride (48 cols)
  __shared__ float  rr[kNNEI][3];
  __shared__ float2 wb0[24];
  __shared__ float  xlds[8][16];
  __shared__ float  h1lds[8][16 * H1S];
  __shared__ float  accw[8][kNG];

  const int a    = blockIdx.x;
  const int tid  = threadIdx.x;
  const int wv   = tid >> 5;
  const int lane = tid & 31;
  const int half = lane >> 4;                // K-half selector (A/B frags)
  const int ln   = lane & 15;                // row (A) / column (B,C,D)

  if (tid < kNNEI) {
    rr[tid][0] = dm[(a * kNNEI + tid) * 4 + 1];
    rr[tid][1] = dm[(a * kNNEI + tid) * 4 + 2];
    rr[tid][2] = dm[(a * kNNEI + tid) * 4 + 3];
  }

  float racc[6] = {0.f, 0.f, 0.f, 0.f, 0.f, 0.f};

  for (int ph = 0; ph < 3; ++ph) {
    const int   idx  = (ph == 0) ? 0 : ((ph == 1) ? 2 : 3);
    const int   Jb   = (ph == 2) ? 20 : 0;
    const int   Kb   = (ph == 0) ? 0 : 20;
    const int   Kn   = (ph == 0) ? 20 : 40;
    const int   np   = 400 << ph;            // 400, 800, 1600
    const float invn = 1.0f / (float)(400 << ph);

    __syncthreads();                         // protect wb0 across phases
    if (tid < 24) wb0[tid] = make_float2(w0g[idx * 24 + tid], b0g[idx * 24 + tid]);
    __syncthreads();

    // per-lane phase constants (biases, layer-1 residual taps)
    float b1n[3], b2n[6], w0r[3], b0r[3];
#pragma unroll
    for (int nt = 0; nt < 3; ++nt) {
      b1n[nt] = b1g[idx * 48 + nt * 16 + ln];
      int km  = (nt * 16 + ln) % 24;
      w0r[nt] = w0g[idx * 24 + km];
      b0r[nt] = b0g[idx * 24 + km];
    }
#pragma unroll
    for (int nt = 0; nt < 6; ++nt) b2n[nt] = b2g[idx * kNG + nt * 16 + ln];

    // Register-resident B fragments (built once per phase, reused all tiles)
    v16h B1[3], B2[12];
#pragma unroll
    for (int nt = 0; nt < 3; ++nt) {
#pragma unroll
      for (int e = 0; e < 16; ++e) {
        int k   = ((e >> 3) << 4) + half * 8 + (e & 7);             // K index, pad 24 -> 32
        float v = (k < 24) ? w1g[idx * 24 * 48 + k * 48 + nt * 16 + ln] : 0.f;
        B1[nt][e] = (_Float16)v;
      }
    }
#pragma unroll
    for (int nt = 0; nt < 6; ++nt) {
#pragma unroll
      for (int s = 0; s < 2; ++s) {
#pragma unroll
        for (int e = 0; e < 16; ++e) {
          int k   = s * 32 + ((e >> 3) << 4) + half * 8 + (e & 7);  // K index, pad 48 -> 64
          float v = (k < 48) ? w2g[idx * 48 * kNG + k * kNG + nt * 16 + ln] : 0.f;
          B2[nt * 2 + s][e] = (_Float16)v;
        }
      }
    }

    const int ntiles = np >> 4;
    for (int t = wv; t < ntiles; t += 8) {
      // x for this wave's 16 pairs
      if (half == 0) {
        int p = t * 16 + ln;
        int j = Jb + p / Kn;
        int k = Kb + p % Kn;
        xlds[wv][ln] = rr[j][0] * rr[k][0] + rr[j][1] * rr[k][1] + rr[j][2] * rr[k][2];
      }
      __builtin_amdgcn_wave_barrier();       // same-wave LDS is in-order
      float xA = xlds[wv][ln];               // x of my A-matrix row
      float xr[8];
#pragma unroll
      for (int r = 0; r < 8; ++r) xr[r] = xlds[wv][r + half * 8];  // C/D rows

      // layer 0 -> A1 fragment (16x32 f16, K=24 padded)
      v16h A1;
#pragma unroll
      for (int e = 0; e < 16; ++e) {
        int k = ((e >> 3) << 4) + half * 8 + (e & 7);
        float v = 0.f;
        if (k < 24) { float2 wb = wb0[k]; v = fast_tanh(xA * wb.x + wb.y); }
        A1[e] = (_Float16)v;
      }

      // layer 1: 16x48 = 3 WMMA, then bias+tanh+residual, stage to LDS
#pragma unroll
      for (int nt = 0; nt < 3; ++nt) {
        v8f c = {0.f, 0.f, 0.f, 0.f, 0.f, 0.f, 0.f, 0.f};
        c = __builtin_amdgcn_wmma_f32_16x16x32_f16(false, A1, false, B1[nt],
                                                   (short)0, c, false, false);
#pragma unroll
        for (int r = 0; r < 8; ++r) {
          float h0res = fast_tanh(xr[r] * w0r[nt] + b0r[nt]);
          float h1v   = fast_tanh(c[r] + b1n[nt]) + h0res;
          h1lds[wv][(r + half * 8) * H1S + nt * 16 + ln] = h1v;
        }
      }
      __builtin_amdgcn_wave_barrier();

      // gather layer-2 A fragments (16x64 f16, K=48 padded, 2 k-steps)
      v16h A2[2];
#pragma unroll
      for (int s = 0; s < 2; ++s) {
#pragma unroll
        for (int e = 0; e < 16; ++e) {
          int k = s * 32 + ((e >> 3) << 4) + half * 8 + (e & 7);
          A2[s][e] = (k < 48) ? (_Float16)h1lds[wv][ln * H1S + k] : (_Float16)0.f;
        }
      }

      // layer 2: 16x96 = 6 N-tiles x 2 WMMA, fuse bias+tanh+residual+reduce
#pragma unroll
      for (int nt = 0; nt < 6; ++nt) {
        v8f c = {0.f, 0.f, 0.f, 0.f, 0.f, 0.f, 0.f, 0.f};
        c = __builtin_amdgcn_wmma_f32_16x16x32_f16(false, A2[0], false, B2[nt * 2 + 0],
                                                   (short)0, c, false, false);
        c = __builtin_amdgcn_wmma_f32_16x16x32_f16(false, A2[1], false, B2[nt * 2 + 1],
                                                   (short)0, c, false, false);
        int nres = (nt % 3) * 16 + ln;       // (nt*16+ln) % 48
        float sacc = 0.f;
#pragma unroll
        for (int r = 0; r < 8; ++r) {
          float h2 = fast_tanh(c[r] + b2n[nt]) + h1lds[wv][(r + half * 8) * H1S + nres];
          sacc += xr[r] * h2;
        }
        sacc += __shfl_xor(sacc, 16, 32);    // fold lane halves (rows 8..15)
        racc[nt] += sacc * invn;             // valid on half==0 lanes
      }
    }
  }

  if (half == 0) {
#pragma unroll
    for (int nt = 0; nt < 6; ++nt) accw[wv][nt * 16 + ln] = racc[nt];
  }
  __syncthreads();
  for (int c = tid; c < kNG; c += 256) {
    float s = 0.f;
#pragma unroll
    for (int w = 0; w < 8; ++w) s += accw[w][c];
    out[a * kNG + c] = s;
  }
}

// -------------------------------------------------------------------------
extern "C" void kernel_launch(void* const* d_in, const int* in_sizes, int n_in,
                              void* d_out, int out_size, void* d_ws, size_t ws_size,
                              hipStream_t stream) {
  (void)in_sizes; (void)n_in; (void)out_size; (void)ws_size;
  const int*   nlist  = (const int*)  d_in[0];
  const float* coord  = (const float*)d_in[1];
  const int*   atype  = (const int*)  d_in[2];
  const float* mean   = (const float*)d_in[3];
  const float* stddev = (const float*)d_in[4];
  const float* w0     = (const float*)d_in[5];
  const float* b0     = (const float*)d_in[6];
  const float* w1     = (const float*)d_in[7];
  const float* b1     = (const float*)d_in[8];
  const float* w2     = (const float*)d_in[9];
  const float* b2     = (const float*)d_in[10];

  float* out_res = (float*)d_out;            // (1, 512, 96)
  float* out_sw  = out_res + kNLOC * kNG;    // (1, 512, 60)
  float* dm      = (float*)d_ws;             // (512, 60, 4) scratch

  env_mat_kernel<<<(kNLOC * kNNEI + 255) / 256, 256, 0, stream>>>(
      nlist, coord, atype, mean, stddev, dm, out_sw);
  three_body_kernel<<<kNLOC, 256, 0, stream>>>(
      dm, w0, b0, w1, b1, w2, b2, out_res);
}
